// FlashAttention_35433480192026
// MI455X (gfx1250) — compile-verified, matrix operand order checked by reading, not run on hardware
//
#include <hip/hip_runtime.h>
#include <hip/hip_bf16.h>

// CDNA5 (gfx1250) types
typedef __attribute__((ext_vector_type(16))) _Float16 v16h;
typedef __attribute__((ext_vector_type(8)))  _Float16 v8h;
typedef __attribute__((ext_vector_type(8)))  float    v8f;
typedef __attribute__((ext_vector_type(4)))  unsigned int v4u;
typedef __attribute__((ext_vector_type(8)))  int v8i;
typedef __attribute__((ext_vector_type(4)))  int v4i;

#if __has_builtin(__builtin_amdgcn_tensor_load_to_lds)
#define HAVE_TDM 1
#else
#define HAVE_TDM 0
#endif

// ---------------------------------------------------------------------------
// Fragment loader per CDNA5 ISA 7.12.2 (wave32), vectorized.
// Works for:
//   A (16x32 MxK) from row-major [M][K] tiles, and
//   B (32x16 KxN) from N-major [N][K] tiles (per-lane K runs contiguous).
// lanes 0-15 -> row/col = lane, kb=0; lanes 16-31 -> row/col = lane-16, kb=8.
// Two 16-byte loads per lane.
// ---------------------------------------------------------------------------
__device__ __forceinline__ v16h load_frag_f16(const _Float16* src, int ld) {
  const int lane = threadIdx.x & 31;
  const int row  = lane & 15;
  const int kb   = (lane >> 4) << 3;
  const _Float16* p = src + row * ld + kb;
  v8h lo = *(const v8h*)(p);
  v8h hi = *(const v8h*)(p + 16);
  v16h f;
#pragma unroll
  for (int i = 0; i < 8; ++i) { f[i] = lo[i]; f[i + 8] = hi[i]; }
  return f;
}

#define WMMA_F16(a, b, c) \
  __builtin_amdgcn_wmma_f32_16x16x32_f16(false, (a), false, (b), (short)0, (c), false, false)

// Fast SiLU: x * rcp(1+exp(-x)) — v_rcp_f32 instead of the IEEE divide expansion.
__device__ __forceinline__ float silu_fast(float x) {
  return x * __builtin_amdgcn_rcpf(1.f + __expf(-x));
}

// ---------------------------------------------------------------------------
// Tensor Data Mover: 2-D f16 tile load, descriptor per ISA 8.3-8.5.
// dims/tiles in elements; addresses in bytes. Not in a cluster -> mask = 0.
// ---------------------------------------------------------------------------
#if HAVE_TDM
__device__ __forceinline__ void tdm_load_2d_f16(unsigned lds_addr, const void* gptr,
                                                unsigned dim0, unsigned dim1,
                                                unsigned stride0,
                                                unsigned tile0, unsigned tile1) {
  const unsigned long long ga = (unsigned long long)gptr;
  v4u g0;
  g0[0] = 1u;                                            // count=1, user D#
  g0[1] = lds_addr;                                      // LDS byte address
  g0[2] = (unsigned)(ga & 0xFFFFFFFFu);                  // global addr lo
  g0[3] = (unsigned)((ga >> 32) & 0x01FFFFFFu) | (2u << 30);  // addr hi | type=2
  v8i g1;
  g1[0] = (int)(1u << 16);                               // data_size=1 (2B), mask=0
  g1[1] = (int)(dim0 << 16);                             // tensor_dim0[15:0]
  g1[2] = (int)((dim0 >> 16) | (dim1 << 16));            // dim0 hi | dim1 lo
  g1[3] = (int)((dim1 >> 16) | (tile0 << 16));           // dim1 hi | tile_dim0
  g1[4] = (int)(tile1 & 0xFFFFu);                        // tile_dim1 (tile_dim2=0)
  g1[5] = (int)stride0;                                  // tensor_dim0_stride lo
  g1[6] = 0;                                             // stride0 hi | dim1_stride lo
  g1[7] = 0;
  v4i z = {0, 0, 0, 0};
#if defined(__clang_major__) && (__clang_major__ >= 23)
  v8i z8 = {0, 0, 0, 0, 0, 0, 0, 0};
  __builtin_amdgcn_tensor_load_to_lds(g0, g1, z, z, z8, 0);
#else
  __builtin_amdgcn_tensor_load_to_lds(g0, g1, z, z, 0);
#endif
}
__device__ __forceinline__ unsigned lds_addr_of(const void* p) {
  return (unsigned)(unsigned long long)p;   // LDS aperture: offset = addr[31:0]
}
#endif

// ---------------------------------------------------------------------------
// Kernel 0: pack weights transposed to f16 so all B-fragments are contiguous.
//   WuT/WvT: [e][d] (512x256), WbT: [s][d] (128x256), WoT: [c][e] (256x512)
// ---------------------------------------------------------------------------
__global__ __launch_bounds__(256)
void pack_kernel(const float* __restrict__ Wu, const float* __restrict__ Wv,
                 const float* __restrict__ Wbase, const float* __restrict__ Wo,
                 _Float16* __restrict__ WuT, _Float16* __restrict__ WvT,
                 _Float16* __restrict__ WbT, _Float16* __restrict__ WoT) {
  const int i = blockIdx.x * 256 + threadIdx.x;
  if (i < 512 * 256) {                 // WuT / WvT
    const int e = i >> 8, d = i & 255;
    WuT[i] = (_Float16)Wu[d * 512 + e];
    WvT[i] = (_Float16)Wv[d * 512 + e];
  }
  if (i < 128 * 256) {                 // WbT
    const int s = i >> 8, d = i & 255;
    WbT[i] = (_Float16)Wbase[d * 128 + s];
  }
  if (i < 256 * 512) {                 // WoT
    const int c = i >> 9, e = i & 511;
    WoT[i] = (_Float16)Wo[e * 256 + c];
  }
}

// ---------------------------------------------------------------------------
// Kernel 1: LayerNorm + (Wu, Wv, Wbase) projections + SiLU + RoPE.
// One block = 16 flat rows of (B*N, D). 8 waves.
// Writes: u row-major f16, v TRANSPOSED f16 [b][e][n], q/k row-major f16.
// ---------------------------------------------------------------------------
__global__ __launch_bounds__(256)
void proj_kernel(const float* __restrict__ query,
                 const float* __restrict__ ln_w, const float* __restrict__ ln_b,
                 const _Float16* __restrict__ WuT, const float* __restrict__ bu,
                 const _Float16* __restrict__ WvT, const float* __restrict__ bv,
                 const _Float16* __restrict__ WbT, const float* __restrict__ bbase,
                 const float* __restrict__ q_w,    const float* __restrict__ q_b,
                 const float* __restrict__ k_w,    const float* __restrict__ k_b,
                 _Float16* __restrict__ u_ws, _Float16* __restrict__ vT_ws,
                 _Float16* __restrict__ q_ws, _Float16* __restrict__ k_ws)
{
  __shared__ __align__(16) _Float16 xn[16][264];   // normalized x (264*2 = 16B-mult)
  __shared__ float qpre[16][128];
  __shared__ float kpre[16][128];

  const int rb   = blockIdx.x;
  const int wv   = threadIdx.x >> 5;
  const int lane = threadIdx.x & 31;

  // ---- LayerNorm (each wave: 2 rows, wave32 shuffle reduction) ----------
#pragma unroll
  for (int rr = 0; rr < 2; ++rr) {
    const int m = wv * 2 + rr;
    const float* xr = query + (size_t)(rb * 16 + m) * 256;
    float xv[8], s1 = 0.f, s2 = 0.f;
#pragma unroll
    for (int i = 0; i < 8; ++i) {
      xv[i] = xr[lane + i * 32];
      s1 += xv[i]; s2 += xv[i] * xv[i];
    }
#pragma unroll
    for (int off = 16; off > 0; off >>= 1) {
      s1 += __shfl_xor(s1, off, 32);
      s2 += __shfl_xor(s2, off, 32);
    }
    const float mu   = s1 * (1.f / 256.f);
    const float var  = s2 * (1.f / 256.f) - mu * mu;
    const float rstd = rsqrtf(var + 1e-5f);
#pragma unroll
    for (int i = 0; i < 8; ++i) {
      const int d = lane + i * 32;
      xn[m][d] = (_Float16)((xv[i] - mu) * rstd * ln_w[d] + ln_b[d]);
    }
  }
  __syncthreads();

  // ---- 72 WMMA tiles: u(32) | v(32) | base(8), K=256 in 8 steps ---------
  for (int t = wv; t < 72; t += 8) {
    const _Float16* WT; const float* bias; int colbase, which;
    if (t < 32)      { WT = WuT; bias = bu;    colbase = t * 16;        which = 0; }
    else if (t < 64) { WT = WvT; bias = bv;    colbase = (t - 32) * 16; which = 1; }
    else             { WT = WbT; bias = bbase; colbase = (t - 64) * 16; which = 2; }

    v8f c = {};
#pragma unroll
    for (int kc = 0; kc < 8; ++kc) {
      v16h a  = load_frag_f16(&xn[0][kc * 32], 264);
      v16h bf = load_frag_f16(WT + (size_t)colbase * 256 + kc * 32, 256);
      c = WMMA_F16(a, bf, c);
    }
#pragma unroll
    for (int r = 0; r < 8; ++r) {
      const int m   = r + ((lane >> 4) << 3);
      const int col = colbase + (lane & 15);
      const int row = rb * 16 + m;
      const float val = silu_fast(c[r] + bias[col]);
      if (which == 0) {
        u_ws[(size_t)row * 512 + col] = (_Float16)val;
      } else if (which == 1) {
        const int bb = row >> 12, n = row & 4095;             // transpose store
        vT_ws[((size_t)bb * 512 + col) * 4096 + n] = (_Float16)val;
      } else {
        qpre[m][col] = val * q_w[col] + q_b[col];
        kpre[m][col] = val * k_w[col] + k_b[col];
      }
    }
  }
  __syncthreads();

  // ---- RoPE (half = 64) -------------------------------------------------
  for (int idx = threadIdx.x; idx < 16 * 64; idx += 256) {
    const int m = idx >> 6, cdim = idx & 63;
    const int row = rb * 16 + m;
    const int pos = row & 4095;
    const float ang = (float)pos * __expf(-(float)cdim * 0.14391157f); // ln(1e4)/64
    float sn, cs;
    __sincosf(ang, &sn, &cs);
    {
      const float x1 = qpre[m][cdim], x2 = qpre[m][cdim + 64];
      q_ws[(size_t)row * 128 + cdim]      = (_Float16)(x1 * cs - x2 * sn);
      q_ws[(size_t)row * 128 + 64 + cdim] = (_Float16)(x2 * cs + x1 * sn);
    }
    {
      const float x1 = kpre[m][cdim], x2 = kpre[m][cdim + 64];
      k_ws[(size_t)row * 128 + cdim]      = (_Float16)(x1 * cs - x2 * sn);
      k_ws[(size_t)row * 128 + 64 + cdim] = (_Float16)(x2 * cs + x1 * sn);
    }
  }
}

// ---------------------------------------------------------------------------
// Kernel 2: flash-style attention with TDM double-buffered tiles.
//   score = relu(q.k^T/4096 + w_rel)^2 ; t = u * (score @ v)
// Grid (256 qtiles, B, 2 e-slabs); 8 waves; wave owns 32 e-cols.
// Waves 0/1 compute the two 16x16 score sub-tiles once, shared via LDS.
// ---------------------------------------------------------------------------
__global__ __launch_bounds__(256)
void attn_kernel(const _Float16* __restrict__ q_ws, const _Float16* __restrict__ k_ws,
                 const _Float16* __restrict__ vT_ws, const _Float16* __restrict__ u_ws,
                 const float* __restrict__ w_rel, _Float16* __restrict__ t_ws)
{
  __shared__ __align__(16) _Float16 kbuf[2][32][128];    // 16 KB, keys x feat
  __shared__ __align__(16) _Float16 vbufT[2][256][32];   // 32 KB, e-slab x keys
  __shared__ __align__(16) _Float16 kern[16][32];        // score A-tile, 1 KB

  const int qt = blockIdx.x, b = blockIdx.y, z = blockIdx.z;
  const int wv = threadIdx.x >> 5, lane = threadIdx.x & 31;
  const int qrow0 = b * 4096 + qt * 16;
  const int ebase = z * 256 + wv * 32;

  // q A-fragments (K=128 in 4 chunks) stay in VGPRs.
  v16h aq[4];
#pragma unroll
  for (int kc = 0; kc < 4; ++kc)
    aq[kc] = load_frag_f16(q_ws + (size_t)qrow0 * 128 + kc * 32, 128);

  const _Float16* kbase  = k_ws + (size_t)b * 4096 * 128;
  const _Float16* vTslab = vT_ws + ((size_t)b * 512 + z * 256) * 4096;

#if HAVE_TDM
  if (wv == 0) {   // prologue: fill buffer 0
    tdm_load_2d_f16(lds_addr_of(&kbuf[0][0][0]),  kbase, 128, 4096, 128, 128, 32);
    tdm_load_2d_f16(lds_addr_of(&vbufT[0][0][0]), vTslab, 4096, 512, 4096, 32, 256);
  }
#endif

  v8f acc[2] = {};

  for (int jp = 0; jp < 128; ++jp) {          // 32 keys / iteration
    const int buf = jp & 1;
    const int key0 = jp * 32;

#if HAVE_TDM
    if (wv == 0) __builtin_amdgcn_s_wait_tensorcnt(0);
    __syncthreads();                          // publish TDM tile to all waves
    if (wv == 0 && jp + 1 < 128) {            // prefetch next tile pair
      const int nk = (jp + 1) * 32;
      tdm_load_2d_f16(lds_addr_of(&kbuf[buf ^ 1][0][0]),
                      kbase + (size_t)nk * 128, 128, 4096, 128, 128, 32);
      tdm_load_2d_f16(lds_addr_of(&vbufT[buf ^ 1][0][0]),
                      vTslab + nk, 4096, 512, 4096, 32, 256);
    }
#else
    {  // cooperative vector copy fallback
      const v8h* gk = (const v8h*)(kbase + (size_t)key0 * 128);
      v8h* dk = (v8h*)&kbuf[buf][0][0];
      for (int i = threadIdx.x; i < 512; i += 256) dk[i] = gk[i];
      for (int i = threadIdx.x; i < 256 * 4; i += 256) {
        const int e = i >> 2, seg = i & 3;
        *(v8h*)&vbufT[buf][e][seg * 8] =
            *(const v8h*)(vTslab + (size_t)e * 4096 + key0 + seg * 8);
      }
      __syncthreads();
    }
#endif

    // ---- score sub-tiles (waves 0,1), shared via LDS --------------------
    if (wv < 2) {
      const int sub = wv;
      v8f c = {};
#pragma unroll
      for (int kc = 0; kc < 4; ++kc) {
        v16h bk = load_frag_f16(&kbuf[buf][sub * 16][kc * 32], 128);
        c = WMMA_F16(aq[kc], bk, c);
      }
#pragma unroll
      for (int r = 0; r < 8; ++r) {
        const int m  = r + ((lane >> 4) << 3);
        const int nn = lane & 15;
        const int qi = qt * 16 + m;
        const int ki = key0 + sub * 16 + nn;
        float val = c[r] * (1.f / 4096.f) + w_rel[4095 + ki - qi];
        val = fmaxf(val, 0.f);
        kern[m][sub * 16 + nn] = (_Float16)(val * val);
      }
    }
    __syncthreads();                          // score tile ready

    // ---- PV: acc += score(16x32) @ v(32 x 32-cols) ----------------------
    v16h ak = load_frag_f16(&kern[0][0], 32);
#pragma unroll
    for (int t = 0; t < 2; ++t) {
      v16h bvf = load_frag_f16(&vbufT[buf][wv * 32 + t * 16][0], 32);
      acc[t] = WMMA_F16(ak, bvf, acc[t]);
    }
  }

  // ---- fused u-gate + store t -------------------------------------------
#pragma unroll
  for (int t = 0; t < 2; ++t) {
#pragma unroll
    for (int r = 0; r < 8; ++r) {
      const int m = r + ((lane >> 4) << 3);
      const int e = ebase + t * 16 + (lane & 15);
      const size_t row = (size_t)(qrow0 + m);
      t_ws[row * 512 + e] = (_Float16)(acc[t][r] * (float)u_ws[row * 512 + e]);
    }
  }
}

// ---------------------------------------------------------------------------
// Kernel 3: out = t @ Wo + bo   ((8192x512)@(512x256), f32 out)
// ---------------------------------------------------------------------------
__global__ __launch_bounds__(256)
void out_kernel(const _Float16* __restrict__ t_ws, const _Float16* __restrict__ WoT,
                const float* __restrict__ bo, float* __restrict__ out)
{
  const int rb = blockIdx.x, wv = threadIdx.x >> 5, lane = threadIdx.x & 31;
  const int colbase = wv * 32;

  v8f acc[2] = {};
#pragma unroll 4
  for (int kc = 0; kc < 16; ++kc) {           // K=512 in 16 steps
    v16h a = load_frag_f16(t_ws + (size_t)(rb * 16) * 512 + kc * 32, 512);
#pragma unroll
    for (int t = 0; t < 2; ++t) {
      v16h bf = load_frag_f16(WoT + (size_t)(colbase + t * 16) * 512 + kc * 32, 512);
      acc[t] = WMMA_F16(a, bf, acc[t]);
    }
  }
#pragma unroll
  for (int t = 0; t < 2; ++t)
#pragma unroll
    for (int r = 0; r < 8; ++r) {
      const int m   = r + ((lane >> 4) << 3);
      const int col = colbase + t * 16 + (lane & 15);
      out[(size_t)(rb * 16 + m) * 256 + col] = acc[t][r] + bo[col];
    }
}

// ---------------------------------------------------------------------------
extern "C" void kernel_launch(void* const* d_in, const int* in_sizes, int n_in,
                              void* d_out, int out_size, void* d_ws, size_t ws_size,
                              hipStream_t stream) {
  const float* query = (const float*)d_in[0];
  const float* ln_w  = (const float*)d_in[1];
  const float* ln_b  = (const float*)d_in[2];
  const float* Wu    = (const float*)d_in[3];
  const float* bu    = (const float*)d_in[4];
  const float* Wv    = (const float*)d_in[5];
  const float* bv    = (const float*)d_in[6];
  const float* Wbase = (const float*)d_in[7];
  const float* bbase = (const float*)d_in[8];
  const float* q_w   = (const float*)d_in[9];
  const float* q_b   = (const float*)d_in[10];
  const float* k_w   = (const float*)d_in[11];
  const float* k_b   = (const float*)d_in[12];
  const float* w_rel = (const float*)d_in[13];
  const float* Wo    = (const float*)d_in[14];
  const float* bo    = (const float*)d_in[15];
  float* out = (float*)d_out;

  // Workspace layout (f16), ~28.8 MB total — fits in the 192 MB L2.
  char* ws = (char*)d_ws;
  _Float16* u_ws  = (_Float16*)(ws + 0);          // 8192*512*2 = 8 MB
  _Float16* vT_ws = (_Float16*)(ws + 8388608);    // 8 MB (transposed [b][e][n])
  _Float16* q_ws  = (_Float16*)(ws + 16777216);   // 2 MB
  _Float16* k_ws  = (_Float16*)(ws + 18874368);   // 2 MB
  _Float16* t_ws  = (_Float16*)(ws + 20971520);   // 8 MB
  _Float16* WuT   = (_Float16*)(ws + 29360128);   // 256 KB
  _Float16* WvT   = (_Float16*)(ws + 29622272);   // 256 KB
  _Float16* WbT   = (_Float16*)(ws + 29884416);   // 64 KB
  _Float16* WoT   = (_Float16*)(ws + 29949952);   // 256 KB

  pack_kernel<<<512, 256, 0, stream>>>(Wu, Wv, Wbase, Wo, WuT, WvT, WbT, WoT);
  proj_kernel<<<512, 256, 0, stream>>>(query, ln_w, ln_b, WuT, bu, WvT, bv,
                                       WbT, bbase, q_w, q_b, k_w, k_b,
                                       u_ws, vT_ws, q_ws, k_ws);
  attn_kernel<<<dim3(256, 2, 2), 256, 0, stream>>>(q_ws, k_ws, vT_ws, u_ws, w_rel, t_ws);
  out_kernel<<<512, 256, 0, stream>>>(t_ws, WoT, bo, out);
}
